// VectorQuantize_87969520156910
// MI455X (gfx1250) — matrix operand order repaired and checked by reading, not compiled, not run on hardware
//
#include <hip/hip_runtime.h>

typedef __attribute__((ext_vector_type(16))) _Float16 v16h;
typedef __attribute__((ext_vector_type(8)))  _Float16 v8h;
typedef __attribute__((ext_vector_type(4)))  _Float16 v4h;
typedef __attribute__((ext_vector_type(8)))  float    v8f;
typedef __attribute__((ext_vector_type(4)))  float    v4f;

#define VQ_N 262144
#define VQ_E 64
#define VQ_K 512
#define LDS_STRIDE 72        // halves per codebook row in LDS (64 + 8 pad -> conflict-free b128 DS loads)
#define CODES_PER_STAGE 256  // stage half the codebook at a time: ~38 KB LDS

// ---------------------------------------------------------------------------
// Kernel 1: nearest-codebook argmin via f16 WMMA (f32 accumulate).
// Block = 256 threads = 8 waves; each wave computes one 16-row tile vs all 512
// codes. dist proxy = ||c||^2 - 2*x.c (the x^2 term is row-constant).
// ---------------------------------------------------------------------------
__global__ __launch_bounds__(256) void vq_argmin_kernel(
    const float* __restrict__ z, const float* __restrict__ cb,
    int* __restrict__ out_idx)
{
  __shared__ _Float16 cb_lds[CODES_PER_STAGE * LDS_STRIDE];
  __shared__ float    csq_lds[CODES_PER_STAGE];

  const int tid  = threadIdx.x;
  const int wave = tid >> 5;
  const int lane = tid & 31;
  const int hi   = lane >> 4;   // which 16-lane half of the wave
  const int col  = lane & 15;   // matrix column / row-within-tile selector
  const int row_base = (blockIdx.x * 8 + wave) * 16;

  // ---- Build A fragments (16 rows x E=64 as two 16x32 f16 tiles) ----------
  // ISA A layout (16-bit, 16x32): lanes 0-15 hold M=lane; halves j=0..7 are
  // K = 8*hi + j, halves j=8..15 are K = 16 + 8*hi + (j-8) (per K=32 chunk).
  v16h a[2];
  {
    const float* zr = z + (size_t)(row_base + col) * VQ_E;
    #pragma unroll
    for (int kc2 = 0; kc2 < 2; ++kc2) {
      const v4f* p4 = (const v4f*)(zr + kc2 * 32 + hi * 8);
      float tmp[16];
      *(v4f*)&tmp[0]  = p4[0];
      *(v4f*)&tmp[4]  = p4[1];
      *(v4f*)&tmp[8]  = p4[4];   // +16 floats
      *(v4f*)&tmp[12] = p4[5];
      #pragma unroll
      for (int j = 0; j < 16; ++j) a[kc2][j] = (_Float16)tmp[j];
    }
  }

  float best[8];
  int   bidx[8];
  #pragma unroll
  for (int r = 0; r < 8; ++r) { best[r] = 3.4e38f; bidx[r] = 0; }

  for (int half = 0; half < 2; ++half) {
    __syncthreads();
    // ---- stage 256 codebook rows (f32 -> f16) into LDS, plus ||c||^2 ------
    {
      const int gcode = half * CODES_PER_STAGE + tid;
      const v4f* src = (const v4f*)(cb + (size_t)gcode * VQ_E);
      v4h* dst = (v4h*)(&cb_lds[tid * LDS_STRIDE]);
      float ss = 0.f;
      #pragma unroll
      for (int q = 0; q < 16; ++q) {
        v4f v = src[q];
        ss += v.x * v.x + v.y * v.y + v.z * v.z + v.w * v.w;
        v4h h;
        h.x = (_Float16)v.x; h.y = (_Float16)v.y;
        h.z = (_Float16)v.z; h.w = (_Float16)v.w;
        dst[q] = h;
      }
      csq_lds[tid] = ss;
    }
    __syncthreads();

    // ---- 16 code tiles of 16 columns each; 2 WMMA per tile (K=64) ---------
    #pragma unroll 4
    for (int t = 0; t < 16; ++t) {
      const int code_local  = t * 16 + col;
      const int code_global = half * CODES_PER_STAGE + code_local;
      // ISA B layout (16-bit, 32x16): lane holds column; halves j=0..7 are K = 8*hi + j, j=8..15 are K = 16 + 8*hi + (j-8).
      const _Float16* bp = &cb_lds[code_local * LDS_STRIDE];
      v8h b0lo = *(const v8h*)(bp + 8 * hi);
      v8h b0hi = *(const v8h*)(bp + 16 + 8 * hi);
      v8h b1lo = *(const v8h*)(bp + 32 + 8 * hi);
      v8h b1hi = *(const v8h*)(bp + 32 + 16 + 8 * hi);
      v16h b0 = __builtin_shufflevector(b0lo, b0hi, 0,1,2,3,4,5,6,7,8,9,10,11,12,13,14,15);
      v16h b1 = __builtin_shufflevector(b1lo, b1hi, 0,1,2,3,4,5,6,7,8,9,10,11,12,13,14,15);

      v8f acc = {};
      acc = __builtin_amdgcn_wmma_f32_16x16x32_f16(false, a[0], false, b0,
                                                   (short)0, acc, false, false);
      acc = __builtin_amdgcn_wmma_f32_16x16x32_f16(false, a[1], false, b1,
                                                   (short)0, acc, false, false);

      const float cs = csq_lds[code_local];
      // C layout: lane L, VGPR r -> M = r + 8*hi, N = col.
      #pragma unroll
      for (int r = 0; r < 8; ++r) {
        float d = cs - 2.0f * acc[r];
        if (d < best[r]) { best[r] = d; bidx[r] = code_global; }
      }
    }
  }

  // ---- cross-lane argmin reduction within each 16-lane half (wave32) ------
  #pragma unroll
  for (int m = 1; m < 16; m <<= 1) {
    #pragma unroll
    for (int r = 0; r < 8; ++r) {
      float od = __shfl_xor(best[r], m, 32);
      int   oi = __shfl_xor(bidx[r], m, 32);
      if (od < best[r] || (od == best[r] && oi < bidx[r])) {
        best[r] = od; bidx[r] = oi;
      }
    }
  }
  // lane (16*hi + r) owns row (row_base + 8*hi + r)
  #pragma unroll
  for (int r = 0; r < 8; ++r) {
    if (col == r) out_idx[row_base + 8 * hi + r] = bidx[r];
  }
}

// ---------------------------------------------------------------------------
// Kernel 2: gather codebook rows into both outputs (pure bandwidth, float4).
// ---------------------------------------------------------------------------
__global__ __launch_bounds__(256) void vq_gather_kernel(
    const int* __restrict__ idx, const float* __restrict__ cb,
    float* __restrict__ out)
{
  const int i   = blockIdx.x * 256 + threadIdx.x;  // N*16 float4 slots
  const int row = i >> 4;
  const int e   = i & 15;
  const int c   = idx[row];
  const v4f v   = ((const v4f*)cb)[c * 16 + e];
  v4f* o = (v4f*)out;
  o[(size_t)row * 16 + e]                       = v;  // z_q_x
  o[(size_t)VQ_N * 16 + (size_t)row * 16 + e]   = v;  // z_q_x_grd (identical)
}

extern "C" void kernel_launch(void* const* d_in, const int* in_sizes, int n_in,
                              void* d_out, int out_size, void* d_ws, size_t ws_size,
                              hipStream_t stream) {
  (void)in_sizes; (void)n_in; (void)out_size; (void)ws_size;
  const float* z  = (const float*)d_in[0];   // [N, E] f32
  const float* cb = (const float*)d_in[1];   // [K, E] f32
  int* idx = (int*)d_ws;                     // [N] scratch indices

  vq_argmin_kernel<<<VQ_N / 128, 256, 0, stream>>>(z, cb, idx);
  vq_gather_kernel<<<(VQ_N * 16) / 256, 256, 0, stream>>>(idx, cb, (float*)d_out);
}
